// MAB_43843026157946
// MI455X (gfx1250) — compile-verified, matrix operand order checked by reading, not run on hardware
//
#include <hip/hip_runtime.h>
#include <hip/hip_bf16.h>

// ---------------------------------------------------------------------------
// CDNA5 / gfx1250 WMMA transformer block (MAB).
// wave32; matmuls on v_wmma_f32_16x16x32_bf16; TDM (tensor_load_to_lds) used
// for bf16 A-tile staging in the GEMMs when the builtin is available.
// ---------------------------------------------------------------------------

typedef __attribute__((ext_vector_type(16))) __bf16 v16bf;
typedef __attribute__((ext_vector_type(8)))  float  v8f;

#if defined(__has_builtin)
#if __has_builtin(__builtin_amdgcn_tensor_load_to_lds) && \
    __has_builtin(__builtin_amdgcn_s_wait_tensorcnt)
#define HAVE_TDM 1
#endif
#endif
#ifndef HAVE_TDM
#define HAVE_TDM 0
#endif

struct FragBits { uint4 lo, hi; };   // 32 bytes == v16bf

__device__ __forceinline__ unsigned short f32_bf16(float f) {
  unsigned int u = __float_as_uint(f);
  u += 0x7FFFu + ((u >> 16) & 1u);           // round-to-nearest-even
  return (unsigned short)(u >> 16);
}
__device__ __forceinline__ unsigned pack_bf16(float lo, float hi) {
  return (unsigned)f32_bf16(lo) | ((unsigned)f32_bf16(hi) << 16);
}

// 16x32 bf16 operand fragment for v_wmma_f32_16x16x32_bf16 from a row-major
// [row][k] buffer with row stride `stride` bf16 elems (stride*2 % 16 == 0).
// Lane L: row (L&15); k elems 0..7 -> half*8+e, 8..15 -> 16+half*8+(e-8).
__device__ __forceinline__ v16bf load_frag(const unsigned short* base,
                                           int stride, int lane) {
  const int r = lane & 15, hf = lane >> 4;
  const char* p = (const char*)(base + (size_t)r * stride) + hf * 16;
  FragBits fb;
  fb.lo = *(const uint4*)(p);
  fb.hi = *(const uint4*)(p + 32);
  return __builtin_bit_cast(v16bf, fb);
}

__device__ __forceinline__ v8f wmma_bf16(v16bf a, v16bf b, v8f c) {
  return __builtin_amdgcn_wmma_f32_16x16x32_bf16(
      false, a, false, b, (short)0, c, false, false);
}

#define GTM 128
#define GTN 128
#define GTK 64
#define GPAD 72   // padded LDS row: 64 bf16 data + 8 pad (144B, 16B aligned)

#if HAVE_TDM
typedef unsigned int __attribute__((ext_vector_type(4))) u32x4;
typedef int __attribute__((ext_vector_type(4))) i32x4;
typedef int __attribute__((ext_vector_type(8))) i32x8;

// 2D bf16 tile load: 128 rows x 64 elems, row stride lda_elems, LDS row
// padding 4 dwords after every 32 dwords (-> GPAD=72 elem rows).
__device__ __forceinline__ void tdm_load_tile_bf16(const void* gptr,
                                                   unsigned lds_byte,
                                                   int lda_elems) {
  const unsigned long long ga = (unsigned long long)gptr;
  u32x4 g0;
  g0[0] = 1u;                                        // count=1 (valid D#)
  g0[1] = lds_byte;                                  // lds_addr
  g0[2] = (unsigned)(ga & 0xFFFFFFFFu);              // global_addr lo
  g0[3] = (unsigned)((ga >> 32) & 0x1FFFFFFu) | 0x80000000u;  // hi | type=2
  i32x8 g1;
  g1[0] = (int)((1u << 16)        // data_size = 2B
              | (1u << 20)        // pad_enable
              | (4u << 22)        // pad_interval: 32 dwords
              | (3u << 25));      // pad_amount: 4 dwords
  g1[1] = (int)(64u << 16);       // tensor_dim0 = 64 (lo half)
  g1[2] = (int)(128u << 16);      // tensor_dim1 = 128 (lo half)
  g1[3] = (int)(64u << 16);       // tile_dim0 = 64
  g1[4] = 128;                    // tile_dim1 = 128, tile_dim2 = 0
  g1[5] = lda_elems;              // tensor_dim0_stride (data_size units)
  g1[6] = 0;
  g1[7] = 0;
  i32x4 z4 = {0, 0, 0, 0};
#if __clang_major__ >= 23
  i32x8 z8 = {0, 0, 0, 0, 0, 0, 0, 0};
  __builtin_amdgcn_tensor_load_to_lds(g0, g1, z4, z4, z8, 0);
#else
  __builtin_amdgcn_tensor_load_to_lds(g0, g1, z4, z4, 0);
#endif
}
#endif  // HAVE_TDM

// ---------------------------------------------------------------------------
// Tiled GEMM:  C[M,N] = act( A[M,K] @ W[K,N] + bias )
// 256 threads / 8 waves; tile 128x128x64; wave tile 32x64 (2x4 16x16 accs).
// A: f32 (converted) or bf16; bf16 path uses TDM double-buffered staging.
// W: f32, column-wise coalesced reads, packed-bf16 b128 LDS stores in [n][k].
// ---------------------------------------------------------------------------
template <bool ABF16>
__global__ __launch_bounds__(256) void gemm_kernel(
    const void* __restrict__ Av, const float* __restrict__ W,
    const float* __restrict__ bias, float* __restrict__ Cf,
    unsigned short* __restrict__ Cb, int N, int Kd,
    int rows_per_batch, long long batch_stride, int lda, int relu) {
  __shared__ __align__(16) unsigned short As[2][GTM * GPAD];
  __shared__ __align__(16) unsigned short Bs[GTN * GPAD];

  const int tid = threadIdx.x, lane = tid & 31, wave = tid >> 5;
  const int wm = wave >> 1, wn = wave & 1;               // 4x2 wave grid
  const int mb = blockIdx.y * GTM, nb = blockIdx.x * GTN;

  v8f acc[2][4];
#pragma unroll
  for (int i = 0; i < 2; ++i)
#pragma unroll
    for (int j = 0; j < 4; ++j)
#pragma unroll
      for (int e = 0; e < 8; ++e) acc[i][j][e] = 0.0f;

  // staging assignments
  const int ar = tid >> 1;               // A row 0..127
  const int ac0 = (tid & 1) * 32;        // A col chunk (32 elems)
  const long long abase =
      (long long)((mb + ar) / rows_per_batch) * batch_stride +
      (long long)((mb + ar) % rows_per_batch) * lda;
  const int bn = tid & 127;              // W n (contiguous across lanes)
  const int bk0 = (tid >> 7) * 32;       // W k chunk (32 elems)

  float4 aregf[8];
  uint4 aregh[4];
  float breg[32];

  auto loadA = [&](int kt) {
    if (ABF16) {
      const uint4* ap =
          (const uint4*)((const unsigned short*)Av + abase + kt + ac0);
#pragma unroll
      for (int j = 0; j < 4; ++j) aregh[j] = ap[j];
    } else {
      const float4* ap = (const float4*)((const float*)Av + abase + kt + ac0);
#pragma unroll
      for (int j = 0; j < 8; ++j) aregf[j] = ap[j];
    }
  };
  auto storeA = [&](int buf) {
    uint4* dst = (uint4*)&As[buf][ar * GPAD + ac0];
    if (ABF16) {
#pragma unroll
      for (int j = 0; j < 4; ++j) dst[j] = aregh[j];
    } else {
#pragma unroll
      for (int j = 0; j < 4; ++j) {
        uint4 u;
        u.x = pack_bf16(aregf[2 * j].x, aregf[2 * j].y);
        u.y = pack_bf16(aregf[2 * j].z, aregf[2 * j].w);
        u.z = pack_bf16(aregf[2 * j + 1].x, aregf[2 * j + 1].y);
        u.w = pack_bf16(aregf[2 * j + 1].z, aregf[2 * j + 1].w);
        dst[j] = u;
      }
    }
  };
  auto loadB = [&](int kt) {
    const float* wp = W + (long long)(kt + bk0) * N + nb;
#pragma unroll
    for (int j = 0; j < 32; ++j) breg[j] = wp[(long long)j * N];
  };
  auto storeB = [&]() {
    uint4* dst = (uint4*)&Bs[bn * GPAD + bk0];
#pragma unroll
    for (int j = 0; j < 4; ++j) {
      uint4 u;
      u.x = pack_bf16(breg[8 * j + 0], breg[8 * j + 1]);
      u.y = pack_bf16(breg[8 * j + 2], breg[8 * j + 3]);
      u.z = pack_bf16(breg[8 * j + 4], breg[8 * j + 5]);
      u.w = pack_bf16(breg[8 * j + 6], breg[8 * j + 7]);
      dst[j] = u;
    }
  };

  // ---- prologue: prefetch first tiles
  loadB(0);
#if HAVE_TDM
  if (ABF16) {
    if (wave == 0)
      tdm_load_tile_bf16((const unsigned short*)Av + (long long)mb * lda,
                         (unsigned)(unsigned long long)(const void*)&As[0][0],
                         lda);
  } else {
    loadA(0);
  }
#else
  loadA(0);
#endif

  int buf = 0;
  for (int kt = 0; kt < Kd; kt += GTK) {
    const bool more = (kt + GTK) < Kd;
#if HAVE_TDM
    if (ABF16) {
      __builtin_amdgcn_s_wait_tensorcnt(0);   // A tile for `buf` landed in LDS
    } else {
      storeA(buf);
    }
#else
    storeA(buf);
#endif
    storeB();
    __syncthreads();

    if (more) {
      loadB(kt + GTK);
#if HAVE_TDM
      if (ABF16) {
        if (wave == 0)
          tdm_load_tile_bf16(
              (const unsigned short*)Av + (long long)mb * lda + kt + GTK,
              (unsigned)(unsigned long long)(const void*)&As[buf ^ 1][0], lda);
      } else {
        loadA(kt + GTK);
      }
#else
      loadA(kt + GTK);
#endif
    }

#pragma unroll
    for (int kc = 0; kc < GTK; kc += 32) {
      v16bf af[2], bb[4];
#pragma unroll
      for (int i = 0; i < 2; ++i)
        af[i] = load_frag(&As[buf][(wm * 32 + i * 16) * GPAD + kc], GPAD, lane);
#pragma unroll
      for (int j = 0; j < 4; ++j)
        bb[j] = load_frag(&Bs[(wn * 64 + j * 16) * GPAD + kc], GPAD, lane);
#pragma unroll
      for (int i = 0; i < 2; ++i)
#pragma unroll
        for (int j = 0; j < 4; ++j) acc[i][j] = wmma_bf16(af[i], bb[j], acc[i][j]);
    }
    __syncthreads();
    buf ^= 1;
  }

  // epilogue: C/D layout -> lane holds column (lane&15); vgpr e row e + 8*half
  const int r = lane & 15, hf = lane >> 4;
#pragma unroll
  for (int i = 0; i < 2; ++i) {
    const int m0 = mb + wm * 32 + i * 16 + 8 * hf;
#pragma unroll
    for (int j = 0; j < 4; ++j) {
      const int col = nb + wn * 64 + j * 16 + r;
      const float bv = bias ? bias[col] : 0.0f;
#pragma unroll
      for (int e = 0; e < 8; ++e) {
        float v = acc[i][j][e] + bv;
        if (relu) v = v > 0.0f ? v : 0.0f;
        const long long idx = (long long)(m0 + e) * N + col;
        if (Cf) Cf[idx] = v;
        if (Cb) Cb[idx] = f32_bf16(v);
      }
    }
  }
}

// ---------------------------------------------------------------------------
// Flash attention: one workgroup (4 waves, 128 thr) per (b, h, 64-query block).
// Each wave owns 16 query rows; K/V tiles of 32 keys shared in LDS.
// knn mask (256 MB) streamed exactly once, fused into the online softmax.
// ---------------------------------------------------------------------------
#define AH 16
#define AB 4
#define ANQ 1024
#define ANK 1024
#define ADV 1024
#define ADH 64

__global__ __launch_bounds__(128) void attn_kernel(
    const unsigned short* __restrict__ qbf, const unsigned short* __restrict__ kbf,
    const unsigned short* __restrict__ vbf, const float* __restrict__ knn,
    unsigned short* __restrict__ obf) {
  __shared__ __align__(16) unsigned short Ks[32 * GPAD];   // [key][dh 64] pad 72
  __shared__ __align__(16) unsigned short Vs[64 * 40];     // [dh][key 32] pad 40
  __shared__ __align__(16) unsigned short Ps[4][16 * 40];  // per-wave P 16x32

  const int tid = threadIdx.x, lane = tid & 31, wave = tid >> 5;
  const int r = lane & 15, hf = lane >> 4;
  const int wg = blockIdx.x;
  const int qblk = wg & 15;                 // NQ/64 = 16
  const int bh = wg >> 4;
  const int h = bh & (AH - 1);
  const int b = bh >> 4;
  const int q0 = qblk * 64 + wave * 16;

  v16bf qfrag[2];
  {
    const unsigned short* qp = qbf + ((size_t)(b * ANQ + q0)) * ADV + h * ADH;
#pragma unroll
    for (int c = 0; c < 2; ++c) qfrag[c] = load_frag(qp + c * 32, ADV, lane);
  }

  float Mx[8], lsum[8];
  v8f o[4];
#pragma unroll
  for (int e = 0; e < 8; ++e) { Mx[e] = -1e30f; lsum[e] = 0.0f; }
#pragma unroll
  for (int t = 0; t < 4; ++t)
#pragma unroll
    for (int e = 0; e < 8; ++e) o[t][e] = 0.0f;

  const float scale = 0.03125f;  // 1/sqrt(1024)
  const float* mbase = knn + (size_t)(h * AB + b) * ANQ * (size_t)ANK;

  const int skk = tid >> 2;            // staging key 0..31
  const int sc0 = (tid & 3) * 16;      // 16-elem chunk

  for (int kt = 0; kt < ANK; kt += 32) {
    {
      const size_t row = (size_t)(b * ANK + kt + skk) * ADV + h * ADH + sc0;
      const uint4* kp = (const uint4*)(kbf + row);
      uint4* kd = (uint4*)&Ks[skk * GPAD + sc0];
      kd[0] = kp[0];
      kd[1] = kp[1];
      const unsigned short* vp = vbf + row;
#pragma unroll
      for (int j = 0; j < 16; ++j) Vs[(sc0 + j) * 40 + skk] = vp[j];
      if (kt + 32 < ANK)
        __builtin_prefetch(kbf + row + (size_t)32 * ADV, 0, 0);
    }
    __syncthreads();

    // S = Q @ K^T over dh (two k=32 chunks), two 16-key subtiles
    v8f s0, s1;
#pragma unroll
    for (int e = 0; e < 8; ++e) { s0[e] = 0.0f; s1[e] = 0.0f; }
#pragma unroll
    for (int c = 0; c < 2; ++c) {
      v16bf b0 = load_frag(&Ks[0 * GPAD + c * 32], GPAD, lane);
      v16bf b1 = load_frag(&Ks[16 * GPAD + c * 32], GPAD, lane);
      s0 = wmma_bf16(qfrag[c], b0, s0);
      s1 = wmma_bf16(qfrag[c], b1, s1);
    }

    float x0[8], x1[8];
#pragma unroll
    for (int e = 0; e < 8; ++e) {
      const size_t mrow = (size_t)(q0 + e + 8 * hf) * ANK + kt;
      x0[e] = s0[e] * scale + mbase[mrow + r];
      x1[e] = s1[e] * scale + mbase[mrow + 16 + r];
    }
    float cf[8];
#pragma unroll
    for (int e = 0; e < 8; ++e) {
      float tm = fmaxf(x0[e], x1[e]);
#pragma unroll
      for (int d = 1; d < 16; d <<= 1) tm = fmaxf(tm, __shfl_xor(tm, d, 16));
      const float nM = fmaxf(Mx[e], tm);
      cf[e] = __expf(Mx[e] - nM);
      Mx[e] = nM;
      const float p0 = __expf(x0[e] - nM);
      const float p1 = __expf(x1[e] - nM);
      float ps = p0 + p1;
#pragma unroll
      for (int d = 1; d < 16; d <<= 1) ps += __shfl_xor(ps, d, 16);
      lsum[e] = lsum[e] * cf[e] + ps;
      Ps[wave][(e + 8 * hf) * 40 + r] = f32_bf16(p0);
      Ps[wave][(e + 8 * hf) * 40 + 16 + r] = f32_bf16(p1);
    }
#pragma unroll
    for (int t = 0; t < 4; ++t)
#pragma unroll
      for (int e = 0; e < 8; ++e) o[t][e] *= cf[e];

    // O += P(16x32) @ V(32x64)
    v16bf pa = load_frag(&Ps[wave][0], 40, lane);
#pragma unroll
    for (int t = 0; t < 4; ++t) {
      v16bf bv = load_frag(&Vs[(t * 16) * 40], 40, lane);
      o[t] = wmma_bf16(pa, bv, o[t]);
    }
    __syncthreads();
  }

#pragma unroll
  for (int e = 0; e < 8; ++e) {
    const float inv = 1.0f / lsum[e];
    const size_t row = (size_t)(b * ANQ + q0 + e + 8 * hf) * ADV + h * ADH;
#pragma unroll
    for (int t = 0; t < 4; ++t)
      obf[row + t * 16 + r] = f32_bf16(o[t][e] * inv);
  }
}

// ---------------------------------------------------------------------------
// out = LayerNorm(X + Y) * g + beta  (row length 1024, one WG per row)
// ---------------------------------------------------------------------------
__global__ __launch_bounds__(256) void add_ln_kernel(
    const float* __restrict__ X, const float* __restrict__ Y,
    const float* __restrict__ g, const float* __restrict__ be,
    float* __restrict__ Of, unsigned short* __restrict__ Ob, int N) {
  __shared__ float red[256];
  const int row = blockIdx.x, tid = threadIdx.x;
  const float* x = X + (long long)row * N;
  const float* y = Y + (long long)row * N;
  float v[4];
  float s = 0.0f;
#pragma unroll
  for (int i = 0; i < 4; ++i) {
    const int c = tid + i * 256;
    v[i] = x[c] + y[c];
    s += v[i];
  }
  red[tid] = s;
  __syncthreads();
  for (int off = 128; off > 0; off >>= 1) {
    if (tid < off) red[tid] += red[tid + off];
    __syncthreads();
  }
  const float mean = red[0] / (float)N;
  __syncthreads();
  float d = 0.0f;
#pragma unroll
  for (int i = 0; i < 4; ++i) {
    const float t = v[i] - mean;
    d += t * t;
  }
  red[tid] = d;
  __syncthreads();
  for (int off = 128; off > 0; off >>= 1) {
    if (tid < off) red[tid] += red[tid + off];
    __syncthreads();
  }
  const float rstd = rsqrtf(red[0] / (float)N + 1e-5f);
#pragma unroll
  for (int i = 0; i < 4; ++i) {
    const int c = tid + i * 256;
    const float o = (v[i] - mean) * rstd * g[c] + be[c];
    const long long idx = (long long)row * N + c;
    if (Of) Of[idx] = o;
    if (Ob) Ob[idx] = f32_bf16(o);
  }
}

// ---------------------------------------------------------------------------
extern "C" void kernel_launch(void* const* d_in, const int* in_sizes, int n_in,
                              void* d_out, int out_size, void* d_ws,
                              size_t ws_size, hipStream_t stream) {
  (void)in_sizes; (void)n_in; (void)out_size; (void)ws_size;
  const float* Q   = (const float*)d_in[0];
  const float* K   = (const float*)d_in[1];
  const float* KM  = (const float*)d_in[2];
  // d_in[3] = mask_pos (fixed 1024)
  const float* Wq  = (const float*)d_in[4];
  const float* bq  = (const float*)d_in[5];
  const float* Wk  = (const float*)d_in[6];
  const float* bk  = (const float*)d_in[7];
  const float* Wv  = (const float*)d_in[8];
  const float* bv  = (const float*)d_in[9];
  const float* Wo0 = (const float*)d_in[10];
  const float* bo0 = (const float*)d_in[11];
  const float* Wo  = (const float*)d_in[12];
  const float* bo  = (const float*)d_in[13];
  const float* Wo2 = (const float*)d_in[14];
  const float* bo2 = (const float*)d_in[15];
  const float* g0  = (const float*)d_in[16];
  const float* be0 = (const float*)d_in[17];
  const float* g1  = (const float*)d_in[18];
  const float* be1 = (const float*)d_in[19];

  const int Bv = 4, NQ = 1024, NK = 1536, DV = 1024, DI = 4096;
  const int M = Bv * NQ;  // 4096

  char* ws = (char*)d_ws;
  size_t off = 0;
  auto alloc = [&](size_t bytes) -> void* {
    void* p = ws + off;
    off += (bytes + 255) & ~(size_t)255;
    return p;
  };
  float*          qf   = (float*)alloc((size_t)M * DV * 4);
  unsigned short* qb   = (unsigned short*)alloc((size_t)M * DV * 2);
  unsigned short* kb   = (unsigned short*)alloc((size_t)M * DV * 2);
  unsigned short* vb   = (unsigned short*)alloc((size_t)M * DV * 2);
  unsigned short* ob   = (unsigned short*)alloc((size_t)M * DV * 2);
  float*          t0   = (float*)alloc((size_t)M * DV * 4);
  float*          Ofl  = (float*)alloc((size_t)M * DV * 4);
  unsigned short* Obb  = (unsigned short*)alloc((size_t)M * DV * 2);
  unsigned short* ff1b = (unsigned short*)alloc((size_t)M * DI * 2);
  float*          ff2f = (float*)alloc((size_t)M * DV * 4);

  const dim3 blk256(256);
  const dim3 gN1024(DV / GTN, M / GTM);   // (8, 32)
  const dim3 gN4096(DI / GTN, M / GTM);   // (32, 32)

  gemm_kernel<false><<<gN1024, blk256, 0, stream>>>(
      Q, Wq, bq, qf, qb, DV, DV, NQ, (long long)NQ * DV, DV, 0);
  gemm_kernel<false><<<gN1024, blk256, 0, stream>>>(
      K, Wk, bk, nullptr, kb, DV, DV, NQ, (long long)NK * DV, DV, 0);
  gemm_kernel<false><<<gN1024, blk256, 0, stream>>>(
      K, Wv, bv, nullptr, vb, DV, DV, NQ, (long long)NK * DV, DV, 0);

  attn_kernel<<<dim3(Bv * 16 * (NQ / 64)), dim3(128), 0, stream>>>(
      qb, kb, vb, KM, ob);

  gemm_kernel<true><<<gN1024, blk256, 0, stream>>>(
      ob, Wo0, bo0, t0, nullptr, DV, DV, M, 0ll, DV, 0);

  add_ln_kernel<<<dim3(M), blk256, 0, stream>>>(qf, t0, g0, be0, Ofl, Obb, DV);

  gemm_kernel<true><<<gN4096, blk256, 0, stream>>>(
      Obb, Wo, bo, nullptr, ff1b, DI, DV, M, 0ll, DV, 1);
  gemm_kernel<true><<<gN1024, blk256, 0, stream>>>(
      ff1b, Wo2, bo2, ff2f, nullptr, DV, DI, M, 0ll, DI, 0);

  add_ln_kernel<<<dim3(M), blk256, 0, stream>>>(
      Ofl, ff2f, g1, be1, (float*)d_out, nullptr, DV);
}
